// GNNEncoder_58463094833681
// MI455X (gfx1250) — matrix operand order repaired and checked by reading, not compiled, not run on hardware
//
#include <hip/hip_runtime.h>
#include <hip/hip_bf16.h>

// ---------------------------------------------------------------------------
// EGNN encoder for gfx1250 (MI455X), wave32, WMMA bf16 path.
// Shapes (fixed by reference): B=8 L=512 K=30 F=128 P=32 ED=33 NL=4 H=128
// GEMM kernels: 2 waves/block, each wave owns TWO 16-row M-strips; the
// (k-tile, n-tile) step loop is explicitly software-pipelined so the next
// B fragment is in flight while the current one feeds two WMMAs.
// ---------------------------------------------------------------------------

#define Bc   8
#define Lc   512
#define Kc   30
#define Fc   128
#define Pc   32
#define EDc  33
#define NLc  4
#define Hc   128
#define Nn   (Bc * Lc)          // 4096 nodes
#define Ec   (Nn * Kc)          // 122880 edges
#define CIN  (2 * Fc + EDc + 1) // 290  (edge MLP input)
#define CKP  320                // CIN padded to mult of 32 -> 10 k-tiles
#define CKT  10
#define NIN  (Fc + EDc)         // 161  (node MLP input)
#define NKP  192                // padded -> 6 k-tiles
#define NKT1 6
#define NKT2 4                  // 128 / 32

typedef __attribute__((ext_vector_type(16))) __bf16 v16bf;
typedef __attribute__((ext_vector_type(8)))  float  v8f;

// A-fragment swizzle (16x32 bf16, ISA 7.12.2): for row m (0..15) and k (0..31)
// within a k-tile, the owning lane and slot:
//   lane = m + ((k>>3)&1)*16 ;  slot = (k&7) + (k>=16 ? 8 : 0)
// Note: for even k, (k, k+1) map to the same lane at (slot, slot+1).
__device__ __forceinline__ int afrag_lane(int m, int kk) { return m + (((kk >> 3) & 1) << 4); }
__device__ __forceinline__ int afrag_slot(int kk)        { return (kk & 7) + ((kk & 16) ? 8 : 0); }

__device__ __forceinline__ v8f wmma_bf16(v16bf a, v16bf b, v8f c) {
    return __builtin_amdgcn_wmma_f32_16x16x32_bf16(false, a, false, b, (short)0, c, false, false);
}

// ---------------------------------------------------------------------------
// 1) KNN: one thread per center node; stable top-30 by (d2, index).
// ---------------------------------------------------------------------------
__global__ void knn_kernel(const float* __restrict__ p, int* __restrict__ nbrG) {
    int n = blockIdx.x * blockDim.x + threadIdx.x;
    if (n >= Nn) return;
    int b = n / Lc, i = n % Lc;
    float px = p[n * 3 + 0], py = p[n * 3 + 1], pz = p[n * 3 + 2];
    float dl[Kc];
    int   il[Kc];
    for (int k = 0; k < Kc; ++k) { dl[k] = 1e30f; il[k] = 0; }
    for (int j = 0; j < Lc; ++j) {
        if (j == i) continue;
        int g = b * Lc + j;
        float dx = px - p[g * 3 + 0];
        float dy = py - p[g * 3 + 1];
        float dz = pz - p[g * 3 + 2];
        float d2 = dx * dx + dy * dy + dz * dz;
        if (d2 < dl[Kc - 1]) {
            int pos = Kc - 1;
            while (pos > 0 && d2 < dl[pos - 1]) {
                dl[pos] = dl[pos - 1]; il[pos] = il[pos - 1]; --pos;
            }
            dl[pos] = d2; il[pos] = j;
        }
    }
    for (int k = 0; k < Kc; ++k) nbrG[n * Kc + k] = b * Lc + il[k];
}

// ---------------------------------------------------------------------------
// 2) Edge attributes: pair_feat[b,src,dst] ++ dist0; validity mask.
//    mask_res is a jax bool array -> 1 byte per element.
// ---------------------------------------------------------------------------
__global__ void edge_attr_kernel(const float* __restrict__ pair_feat,
                                 const unsigned char* __restrict__ mask,
                                 const float* __restrict__ p,
                                 const int* __restrict__ nbrG,
                                 float* __restrict__ eattr,
                                 float* __restrict__ validB) {
    int e = blockIdx.x * blockDim.x + threadIdx.x;
    if (e >= Ec) return;
    int n    = e / Kc;
    int src  = nbrG[e];
    int b    = n / Lc;
    int dstl = n % Lc;
    int srcl = src % Lc;
    const float* pf = pair_feat + ((size_t)(b * Lc + srcl) * Lc + dstl) * Pc;
    float* out = eattr + (size_t)e * EDc;
    for (int c = 0; c < Pc; ++c) out[c] = pf[c];
    float dx = p[src * 3 + 0] - p[n * 3 + 0];
    float dy = p[src * 3 + 1] - p[n * 3 + 1];
    float dz = p[src * 3 + 2] - p[n * 3 + 2];
    out[Pc] = sqrtf(dx * dx + dy * dy + dz * dz);
    validB[e] = (mask[src] && mask[n]) ? 1.0f : 0.0f;
}

__global__ void agg_kernel(const float* __restrict__ eattr, float* __restrict__ aggE) {
    int idx = blockIdx.x * blockDim.x + threadIdx.x;
    if (idx >= Nn * EDc) return;
    int n = idx / EDc, c = idx % EDc;
    float s = 0.f;
    for (int k = 0; k < Kc; ++k) s += eattr[(size_t)(n * Kc + k) * EDc + c];
    aggE[idx] = s * (1.0f / Kc);
}

// ---------------------------------------------------------------------------
// 3) Weight prep: f32 [K x 128] -> bf16 B-fragments, all NL layers.
//    Fragment layout: out[((t*8+j)*32 + lane)*16 + i] = W[k][n]
//      n = j*16 + (lane&15), k = t*32 + ((lane>>4)&1)*8 + (i&7) + (i>=8 ? 16 : 0)  (0 if k>=realK)
// ---------------------------------------------------------------------------
__global__ void prep_w_kernel(const float* __restrict__ W, unsigned short* __restrict__ outraw,
                              int realK, int KT, int inLayerStride) {
    int total = NLc * KT * 8 * 32 * 16;
    int idx = blockIdx.x * blockDim.x + threadIdx.x;
    if (idx >= total) return;
    int i    = idx & 15;
    int lane = (idx >> 4) & 31;
    int j    = (idx >> 9) & 7;
    int tl   = idx >> 12;             // l*KT + t
    int t    = tl % KT;
    int l    = tl / KT;
    int ncol = j * 16 + (lane & 15);
    int k    = t * 32 + ((lane >> 4) & 1) * 8 + (i & 7) + ((i & 8) << 1);
    float v  = (k < realK) ? W[(size_t)l * inLayerStride + (size_t)k * Hc + ncol] : 0.0f;
    reinterpret_cast<__bf16*>(outraw)[idx] = (__bf16)v;
}

// ---------------------------------------------------------------------------
// 4a) Edge MLP: 64 edges per block, 2 waves, each wave owns strips {2s,2s+1}.
//     cf = [x[src](128) | x[dst](128) | edge_attr(33) | rd(1) | 0-pad]
//     h = relu(cf@W1+b1); w_e = h@w2 + b2  (fused via lane-shuffle reduce)
// ---------------------------------------------------------------------------
__global__ __launch_bounds__(64) void edge_mlp_kernel(
        const float* __restrict__ x, const float* __restrict__ p,
        const int* __restrict__ nbrG, const float* __restrict__ eattr,
        const unsigned short* __restrict__ wBraw,
        const float* __restrict__ b1, const float* __restrict__ w2,
        const float* __restrict__ b2p, float* __restrict__ wbuf) {
    __shared__ __align__(32) __bf16 cf[64 * CKP];   // 40 KB, A-fragment swizzled
    __shared__ int   srcL[64];
    __shared__ float rdL[64];
    const __bf16* wB = reinterpret_cast<const __bf16*>(wBraw);

    int tid = threadIdx.x;          // 0..63
    int ebase = blockIdx.x * 64;

    // Phase 0: per-row metadata (all 64 threads, one row each).
    {
        int e = ebase + tid;
        int n = e / Kc;
        int src = nbrG[e];
        srcL[tid] = src;
        float dx = p[n * 3 + 0] - p[src * 3 + 0];
        float dy = p[n * 3 + 1] - p[src * 3 + 1];
        float dz = p[n * 3 + 2] - p[src * 3 + 2];
        rdL[tid] = sqrtf(dx * dx + dy * dy + dz * dz);
    }
    __syncthreads();

    // Phase 1: stage cf (bf16) into LDS in A-fragment order, 2 k's per iter
    // (even k and k+1 land in adjacent slots of the same lane).
    for (int idx = tid; idx < 64 * (CKP / 2); idx += 64) {
        int r = idx / (CKP / 2);
        int k = (idx - r * (CKP / 2)) * 2;
        int e = ebase + r;
        int n = e / Kc;
        float v0, v1;
        {
            int kk0 = k, kk1 = k + 1;
            v0 = (kk0 < Fc) ? x[srcL[r] * Fc + kk0]
               : (kk0 < 2 * Fc) ? x[n * Fc + (kk0 - Fc)]
               : (kk0 < 2 * Fc + EDc) ? eattr[(size_t)e * EDc + (kk0 - 2 * Fc)]
               : (kk0 == CIN - 1) ? rdL[r] : 0.0f;
            v1 = (kk1 < Fc) ? x[srcL[r] * Fc + kk1]
               : (kk1 < 2 * Fc) ? x[n * Fc + (kk1 - Fc)]
               : (kk1 < 2 * Fc + EDc) ? eattr[(size_t)e * EDc + (kk1 - 2 * Fc)]
               : (kk1 == CIN - 1) ? rdL[r] : 0.0f;
        }
        int t = k >> 5, kk = k & 31, m = r & 15, strip = r >> 4;
        int base = ((strip * CKT + t) * 32 + afrag_lane(m, kk)) * 16 + afrag_slot(kk);
        cf[base]     = (__bf16)v0;
        cf[base + 1] = (__bf16)v1;
    }
    __syncthreads();

    // Phase 2: software-pipelined WMMA over 80 (t,j) steps.
    int lane = tid & 31;
    int s    = tid >> 5;
    int s0 = 2 * s, s1 = 2 * s + 1;

    auto loadA = [&](int strip, int t) {
        return *(const v16bf*)&cf[((strip * CKT + t) * 32 + lane) * 16];
    };
    auto loadB = [&](int step) {
        return *(const v16bf*)&wB[(step * 32 + lane) * 16];
    };

    v8f acc0[8] = {};
    v8f acc1[8] = {};
    v16bf a0 = loadA(s0, 0);
    v16bf a1 = loadA(s1, 0);
    v16bf bb = loadB(0);
#pragma unroll
    for (int step = 0; step < CKT * 8; ++step) {
        int j = step & 7;
        v16bf bn{}, a0n{}, a1n{};
        if (step + 1 < CKT * 8) {
            bn = loadB(step + 1);                      // next B in flight
            if (((step + 1) & 7) == 0) {
                int tn = (step + 1) >> 3;
                a0n = loadA(s0, tn);
                a1n = loadA(s1, tn);
                if (tn + 1 < CKT)                      // warm L0/L2 two tiles out
                    __builtin_prefetch(&wB[((tn + 1) * 8 * 32) * 16], 0, 1);
            }
        }
        acc0[j] = wmma_bf16(a0, bb, acc0[j]);
        acc1[j] = wmma_bf16(a1, bb, acc1[j]);
        bb = bn;
        if (((step + 1) & 7) == 0 && step + 1 < CKT * 8) { a0 = a0n; a1 = a1n; }
    }

    // Epilogue: bias + relu + dot(w2), shuffle-reduced over each lane half.
    int col = lane & 15;
    float p0[8] = {0.f, 0.f, 0.f, 0.f, 0.f, 0.f, 0.f, 0.f};
    float p1[8] = {0.f, 0.f, 0.f, 0.f, 0.f, 0.f, 0.f, 0.f};
#pragma unroll
    for (int j = 0; j < 8; ++j) {
        float bj = b1[j * 16 + col];
        float wj = w2[j * 16 + col];
#pragma unroll
        for (int v = 0; v < 8; ++v) {
            float h0 = acc0[j][v] + bj; h0 = h0 > 0.f ? h0 : 0.f;
            float h1 = acc1[j][v] + bj; h1 = h1 > 0.f ? h1 : 0.f;
            p0[v] += h0 * wj;
            p1[v] += h1 * wj;
        }
    }
#pragma unroll
    for (int off = 1; off < 16; off <<= 1) {
#pragma unroll
        for (int v = 0; v < 8; ++v) {
            p0[v] += __shfl_xor(p0[v], off, 32);
            p1[v] += __shfl_xor(p1[v], off, 32);
        }
    }
    if (col == 0) {
        int half = lane >> 4;
        float b2 = b2p[0];
#pragma unroll
        for (int v = 0; v < 8; ++v) {
            wbuf[ebase + s0 * 16 + half * 8 + v] = p0[v] + b2;
            wbuf[ebase + s1 * 16 + half * 8 + v] = p1[v] + b2;
        }
    }
}

// ---------------------------------------------------------------------------
// 4b) Coordinate update: p_next = p + mean_k(w * (p[dst]-p[src]) * valid)
// ---------------------------------------------------------------------------
__global__ void coord_update_kernel(const float* __restrict__ p,
                                    const int* __restrict__ nbrG,
                                    const float* __restrict__ wbuf,
                                    const float* __restrict__ validB,
                                    float* __restrict__ pout) {
    int n = blockIdx.x * blockDim.x + threadIdx.x;
    if (n >= Nn) return;
    float px = p[n * 3 + 0], py = p[n * 3 + 1], pz = p[n * 3 + 2];
    float ax = 0.f, ay = 0.f, az = 0.f;
    for (int k = 0; k < Kc; ++k) {
        int e = n * Kc + k;
        int src = nbrG[e];
        float w = wbuf[e] * validB[e];
        ax += w * (px - p[src * 3 + 0]);
        ay += w * (py - p[src * 3 + 1]);
        az += w * (pz - p[src * 3 + 2]);
    }
    const float inv = 1.0f / Kc;
    pout[n * 3 + 0] = px + ax * inv;
    pout[n * 3 + 1] = py + ay * inv;
    pout[n * 3 + 2] = pz + az * inv;
}

// ---------------------------------------------------------------------------
// 4c) Node MLP: x_new = relu([x|agg_edge] @ W1 + b1) @ W2 + b2
//     Two chained WMMA GEMMs, relu routed through LDS in A-fragment layout.
//     2 waves/block, each wave owns strips {2s, 2s+1} (32 node rows).
// ---------------------------------------------------------------------------
__global__ __launch_bounds__(64) void node_mlp_kernel(
        const float* __restrict__ x, const float* __restrict__ aggE,
        const unsigned short* __restrict__ w1raw, const float* __restrict__ b1,
        const unsigned short* __restrict__ w2raw, const float* __restrict__ b2,
        float* __restrict__ xout) {
    __shared__ __align__(32) __bf16 nf[64 * NKP];   // 24 KB
    __shared__ __align__(32) __bf16 hs[64 * Fc];    // 16 KB
    const __bf16* w1 = reinterpret_cast<const __bf16*>(w1raw);
    const __bf16* w2 = reinterpret_cast<const __bf16*>(w2raw);

    int tid = threadIdx.x;          // 0..63
    int nbase = blockIdx.x * 64;

    for (int idx = tid; idx < 64 * (NKP / 2); idx += 64) {
        int r = idx / (NKP / 2);
        int k = (idx - r * (NKP / 2)) * 2;
        int n = nbase + r;
        float v0 = (k < Fc) ? x[n * Fc + k]
                 : (k < NIN) ? aggE[n * EDc + (k - Fc)] : 0.0f;
        int k1 = k + 1;
        float v1 = (k1 < Fc) ? x[n * Fc + k1]
                 : (k1 < NIN) ? aggE[n * EDc + (k1 - Fc)] : 0.0f;
        int t = k >> 5, kk = k & 31, m = r & 15, strip = r >> 4;
        int base = ((strip * NKT1 + t) * 32 + afrag_lane(m, kk)) * 16 + afrag_slot(kk);
        nf[base]     = (__bf16)v0;
        nf[base + 1] = (__bf16)v1;
    }
    __syncthreads();

    int lane = tid & 31;
    int s    = tid >> 5;
    int s0 = 2 * s, s1 = 2 * s + 1;
    int col  = lane & 15;
    int half = lane >> 4;

    auto loadA1 = [&](int strip, int t) {
        return *(const v16bf*)&nf[((strip * NKT1 + t) * 32 + lane) * 16];
    };
    auto loadA2 = [&](int strip, int t) {
        return *(const v16bf*)&hs[((strip * NKT2 + t) * 32 + lane) * 16];
    };
    auto loadB1 = [&](int step) {
        return *(const v16bf*)&w1[(step * 32 + lane) * 16];
    };
    auto loadB2 = [&](int step) {
        return *(const v16bf*)&w2[(step * 32 + lane) * 16];
    };

    // GEMM1: 48 software-pipelined steps.
    v8f acc0[8] = {};
    v8f acc1[8] = {};
    {
        v16bf a0 = loadA1(s0, 0);
        v16bf a1 = loadA1(s1, 0);
        v16bf bb = loadB1(0);
#pragma unroll
        for (int step = 0; step < NKT1 * 8; ++step) {
            int j = step & 7;
            v16bf bn{}, a0n{}, a1n{};
            if (step + 1 < NKT1 * 8) {
                bn = loadB1(step + 1);
                if (((step + 1) & 7) == 0) {
                    int tn = (step + 1) >> 3;
                    a0n = loadA1(s0, tn);
                    a1n = loadA1(s1, tn);
                }
            }
            acc0[j] = wmma_bf16(a0, bb, acc0[j]);
            acc1[j] = wmma_bf16(a1, bb, acc1[j]);
            bb = bn;
            if (((step + 1) & 7) == 0 && step + 1 < NKT1 * 8) { a0 = a0n; a1 = a1n; }
        }
    }

    // relu(acc + b1) -> LDS in A-fragment layout for GEMM2 (K = 128).
#pragma unroll
    for (int j = 0; j < 8; ++j) {
        float bj = b1[j * 16 + col];
        int kcol = j * 16 + col;
        int t2 = kcol >> 5, kk2 = kcol & 31;
        int slot = afrag_slot(kk2);
        int lsh  = ((kk2 >> 3) & 1) << 4;
#pragma unroll
        for (int v = 0; v < 8; ++v) {
            int m2 = half * 8 + v;
            float h0 = acc0[j][v] + bj; h0 = h0 > 0.f ? h0 : 0.f;
            float h1 = acc1[j][v] + bj; h1 = h1 > 0.f ? h1 : 0.f;
            hs[((s0 * NKT2 + t2) * 32 + (m2 + lsh)) * 16 + slot] = (__bf16)h0;
            hs[((s1 * NKT2 + t2) * 32 + (m2 + lsh)) * 16 + slot] = (__bf16)h1;
        }
    }
    __syncthreads();

    // GEMM2: 32 software-pipelined steps.
    v8f d0[8] = {};
    v8f d1[8] = {};
    {
        v16bf a0 = loadA2(s0, 0);
        v16bf a1 = loadA2(s1, 0);
        v16bf bb = loadB2(0);
#pragma unroll
        for (int step = 0; step < NKT2 * 8; ++step) {
            int j = step & 7;
            v16bf bn{}, a0n{}, a1n{};
            if (step + 1 < NKT2 * 8) {
                bn = loadB2(step + 1);
                if (((step + 1) & 7) == 0) {
                    int tn = (step + 1) >> 3;
                    a0n = loadA2(s0, tn);
                    a1n = loadA2(s1, tn);
                }
            }
            d0[j] = wmma_bf16(a0, bb, d0[j]);
            d1[j] = wmma_bf16(a1, bb, d1[j]);
            bb = bn;
            if (((step + 1) & 7) == 0 && step + 1 < NKT2 * 8) { a0 = a0n; a1 = a1n; }
        }
    }
#pragma unroll
    for (int j = 0; j < 8; ++j) {
        float bj = b2[j * 16 + col];
#pragma unroll
        for (int v = 0; v < 8; ++v) {
            int r0 = nbase + s0 * 16 + half * 8 + v;
            int r1 = nbase + s1 * 16 + half * 8 + v;
            xout[(size_t)r0 * Fc + j * 16 + col] = d0[j][v] + bj;
            xout[(size_t)r1 * Fc + j * 16 + col] = d1[j][v] + bj;
        }
    }
}

// ---------------------------------------------------------------------------
// Host launch
// ---------------------------------------------------------------------------
extern "C" void kernel_launch(void* const* d_in, const int* in_sizes, int n_in,
                              void* d_out, int out_size, void* d_ws, size_t ws_size,
                              hipStream_t stream) {
    const float* t_in  = (const float*)d_in[1];
    const float* res   = (const float*)d_in[2];
    const float* pairf = (const float*)d_in[3];
    const unsigned char* mask = (const unsigned char*)d_in[4];   // jax bool -> 1 byte
    const float* cw1 = (const float*)d_in[5];
    const float* cb1 = (const float*)d_in[6];
    const float* cw2 = (const float*)d_in[7];
    const float* cb2 = (const float*)d_in[8];
    const float* nw1 = (const float*)d_in[9];
    const float* nb1 = (const float*)d_in[10];
    const float* nw2 = (const float*)d_in[11];
    const float* nb2 = (const float*)d_in[12];

    // Workspace carve-out (256 B aligned), ~23 MB total.
    char* ws = (char*)d_ws;
    size_t off = 0;
    auto carve = [&](size_t bytes) {
        size_t cur = off;
        off += (bytes + 255) & ~(size_t)255;
        return (void*)(ws + cur);
    };
    int*   nbrG   = (int*)  carve((size_t)Ec * 4);
    float* pA     = (float*)carve((size_t)Nn * 3 * 4);
    float* pB     = (float*)carve((size_t)Nn * 3 * 4);
    float* xA     = (float*)carve((size_t)Nn * Fc * 4);
    float* xB     = (float*)carve((size_t)Nn * Fc * 4);
    float* eattr  = (float*)carve((size_t)Ec * EDc * 4);
    float* validB = (float*)carve((size_t)Ec * 4);
    float* aggE   = (float*)carve((size_t)Nn * EDc * 4);
    float* wbuf   = (float*)carve((size_t)Ec * 4);
    unsigned short* wB1 = (unsigned short*)carve((size_t)NLc * CKT  * 8 * 32 * 16 * 2);
    unsigned short* wN1 = (unsigned short*)carve((size_t)NLc * NKT1 * 8 * 32 * 16 * 2);
    unsigned short* wN2 = (unsigned short*)carve((size_t)NLc * NKT2 * 8 * 32 * 16 * 2);

    // Initialize p and x from inputs (device-to-device copies on stream).
    hipMemcpyAsync(pA, t_in, (size_t)Nn * 3 * 4, hipMemcpyDeviceToDevice, stream);
    hipMemcpyAsync(xA, res,  (size_t)Nn * Fc * 4, hipMemcpyDeviceToDevice, stream);

    knn_kernel<<<Nn / 256, 256, 0, stream>>>(pA, nbrG);
    edge_attr_kernel<<<(Ec + 255) / 256, 256, 0, stream>>>(pairf, mask, pA, nbrG, eattr, validB);
    agg_kernel<<<(Nn * EDc + 255) / 256, 256, 0, stream>>>(eattr, aggE);

    // Pre-swizzle all layer weights into bf16 B-fragments.
    prep_w_kernel<<<(NLc * CKT  * 4096 + 255) / 256, 256, 0, stream>>>(cw1, wB1, CIN, CKT,  CIN * Hc);
    prep_w_kernel<<<(NLc * NKT1 * 4096 + 255) / 256, 256, 0, stream>>>(nw1, wN1, NIN, NKT1, NIN * Hc);
    prep_w_kernel<<<(NLc * NKT2 * 4096 + 255) / 256, 256, 0, stream>>>(nw2, wN2, Fc,  NKT2, Fc  * Hc);

    float* pcur = pA; float* pnext = pB;
    float* xcur = xA; float* xnext = xB;
    for (int l = 0; l < NLc; ++l) {
        edge_mlp_kernel<<<Ec / 64, 64, 0, stream>>>(
            xcur, pcur, nbrG, eattr,
            wB1 + (size_t)l * CKT * 8 * 32 * 16,
            cb1 + l * Hc, cw2 + l * Hc, cb2 + l, wbuf);
        coord_update_kernel<<<Nn / 256, 256, 0, stream>>>(pcur, nbrG, wbuf, validB, pnext);
        float* pt = pcur; pcur = pnext; pnext = pt;

        float* outp = (l == NLc - 1) ? (float*)d_out : xnext;
        node_mlp_kernel<<<Nn / 64, 64, 0, stream>>>(
            xcur, aggE,
            wN1 + (size_t)l * NKT1 * 8 * 32 * 16, nb1 + l * Hc,
            wN2 + (size_t)l * NKT2 * 8 * 32 * 16, nb2 + l * Fc,
            outp);
        float* xt = xcur; xcur = outp; xnext = xt;
    }
    (void)in_sizes; (void)n_in; (void)out_size; (void)ws_size;
}
